// HGT_71768903516648
// MI455X (gfx1250) — matrix operand order repaired and checked by reading, not compiled, not run on hardware
//
#include <hip/hip_runtime.h>
#include <math.h>

typedef __attribute__((ext_vector_type(16))) _Float16 v16h;
typedef __attribute__((ext_vector_type(8)))  _Float16 v8h;
typedef __attribute__((ext_vector_type(8)))  float    v8f;

#define NP_  200000
#define NA_  100000
#define EW_  300000
#define EC_  300000
#define HID_ 128
#define H_   4

__device__ __constant__ float kSCALE = 0.17677669529663687f; // 1/sqrt(32)

// Swizzled B index: element (kr, cn) of a [K,N] weight -> position so that the
// 16 f16 values each lane needs for V_WMMA_F32_16X16X32_F16 are contiguous.
// Layout: [ks][nt][lane=khalf*16+n16][kw], ks=kr>>5, khalf=(kr>>4)&1, kw=kr&15.
__device__ inline size_t bswz_idx(int kr, int cn, int NT) {
  int ks = kr >> 5, khalf = (kr >> 4) & 1, kw = kr & 15;
  int nt = cn >> 4, n16 = cn & 15;
  return ((((size_t)ks * NT + nt) * 32) + khalf * 16 + n16) * 16 + kw;
}

// ---------------------------------------------------------------- fill
__global__ void hgt_fill(float* __restrict__ p, float v, int n) {
  int i = blockIdx.x * blockDim.x + threadIdx.x;
  if (i < n) p[i] = v;
}

// ------------------------------------- prep: f32 [K,N] weight -> swizzled f16
__global__ void hgt_prep_b(const float* __restrict__ W, _Float16* __restrict__ Bswz,
                           int K, int N) {
  int i = blockIdx.x * blockDim.x + threadIdx.x;
  if (i >= K * N) return;
  int kr = i / N, cn = i - kr * N;
  Bswz[bswz_idx(kr, cn, N >> 4)] = (_Float16)W[i];
}

// ------------------------------- compose W_eff = W * blockdiag(R) -> swizzled
// R is [4,32,32] (h,d,e); also emits composed bias beff[128] (f32)
__global__ void hgt_compose(const float* __restrict__ W, const float* __restrict__ b,
                            const float* __restrict__ R,
                            _Float16* __restrict__ Weffh, float* __restrict__ beff) {
  int idx = blockIdx.x * blockDim.x + threadIdx.x;
  if (idx >= 129 * 128) return;
  int row = idx >> 7;        // 0..128 (row 128 == bias)
  int col = idx & 127;
  int h = col >> 5, e = col & 31;
  const float* Rh = R + h * 1024;
  float acc = 0.f;
  if (row < 128) {
    const float* wr = W + row * 128 + h * 32;
#pragma unroll
    for (int d = 0; d < 32; ++d) acc += wr[d] * Rh[d * 32 + e];
    Weffh[bswz_idx(row, col, 8)] = (_Float16)acc;
  } else {
    const float* br = b + h * 32;
#pragma unroll
    for (int d = 0; d < 32; ++d) acc += br[d] * Rh[d * 32 + e];
    beff[col] = acc;
  }
}

// ---------------------------------------------------------------- WMMA GEMM
// C[M,N] = post( pre(A)[M,K] @ W[K,N] + bias )  (+ optional gated skip mix)
// LDS-free: A operand loaded directly from global in WMMA lane layout
// (two contiguous 8-float runs per lane); B operand pre-swizzled f16.
// PRE: 0=none, 1=exact GELU on A.  POST: 0=none, 1=relu.
// MIX: C = beta*C + (1-beta)*mixin, beta=sigmoid(*betap).  NTT = N/16.
template <int PRE, int POST, bool MIX, int NTT, typename TOUT>
__global__ __launch_bounds__(256)
void hgt_gemm(const float* __restrict__ A, const _Float16* __restrict__ Bswz,
              const float* __restrict__ bias,
              const float* __restrict__ mixin, const float* __restrict__ betap,
              TOUT* __restrict__ C, int M, int K) {
  constexpr int N = NTT * 16;
  const int tid  = threadIdx.x;
  const int wave = tid >> 5;
  const int lane = tid & 31;
  const int half = lane >> 4;
  const int l16  = lane & 15;
  const int rowBase = blockIdx.x * 128 + wave * 16;

  // clamp OOB rows for loads (stores are guarded; WMMA keeps EXEC all-ones)
  int myrow = rowBase + l16;
  const float* Arow = A + (size_t)((myrow < M) ? myrow : (M - 1)) * K;

  v8f acc[NTT];
#pragma unroll
  for (int i = 0; i < NTT; ++i)
    acc[i] = (v8f){0.f, 0.f, 0.f, 0.f, 0.f, 0.f, 0.f, 0.f};

  for (int k0 = 0; k0 < K; k0 += 32) {
    const int ks = k0 >> 5;
    // A operand: contiguous run at cols k0+half*16 (+0..15)
    const float4* p0 = (const float4*)(Arow + k0 + half * 16);
    const float4* p1 = (const float4*)(Arow + k0 + half * 16 + 8);
    float4 x0 = p0[0], x1 = p0[1], x2 = p1[0], x3 = p1[1];
    float xs[16] = {x0.x, x0.y, x0.z, x0.w, x1.x, x1.y, x1.z, x1.w,
                    x2.x, x2.y, x2.z, x2.w, x3.x, x3.y, x3.z, x3.w};
    v16h a;
#pragma unroll
    for (int j = 0; j < 16; ++j) {
      float x = xs[j];
      if (PRE == 1) x = 0.5f * x * (1.f + erff(x * 0.70710678118f));
      a[j] = (_Float16)x;
    }
#pragma unroll
    for (int nt = 0; nt < NTT; ++nt) {
      const v8h* bp =
          (const v8h*)(Bswz + ((((size_t)ks * NTT + nt) * 32) + lane) * 16);
      v8h lo = bp[0], hi = bp[1];
      v16h b = __builtin_shufflevector(lo, hi, 0, 1, 2, 3, 4, 5, 6, 7,
                                       8, 9, 10, 11, 12, 13, 14, 15);
      acc[nt] = __builtin_amdgcn_wmma_f32_16x16x32_f16(
          false, a, false, b, (short)0, acc[nt], false, false);
    }
  }

  float beta = 0.f, omb = 0.f;
  if (MIX) {
    float s = betap[0];
    beta = 1.f / (1.f + __expf(-s));
    omb  = 1.f - beta;
  }

  // C/D layout: VGPR r -> (M=r, N=lane) lanes 0-15, (M=r+8, N=lane-16) 16-31
#pragma unroll
  for (int nt = 0; nt < NTT; ++nt) {
    int col = nt * 16 + l16;
    float bv = bias[col];
#pragma unroll
    for (int r = 0; r < 8; ++r) {
      int row = rowBase + half * 8 + r;
      if (row < M) {
        float v = acc[nt][r] + bv;
        if (POST == 1) v = fmaxf(v, 0.f);
        if (MIX) v = beta * v + omb * mixin[(size_t)row * N + col];
        C[(size_t)row * N + col] = (TOUT)v;
      }
    }
  }
}

// ---------------------------------------------------------------- edge passes
__device__ inline void atomicMaxF(float* addr, float val) {
  int* ai = reinterpret_cast<int*>(addr);
  int old = *ai;
  while (__int_as_float(old) < val) {
    int assumed = old;
    old = atomicCAS(ai, assumed, __float_as_int(val));
    if (old == assumed) break;
  }
}

// pass A: alpha[e,h] = (q[dst,h,:] . kr[src,h,:]) * p[h] * SCALE ; m = segmax
__global__ void hgt_edge_alpha(const _Float16* __restrict__ q,
                               const _Float16* __restrict__ kr,
                               const int* __restrict__ src, const int* __restrict__ dst,
                               const float* __restrict__ prel,
                               float* __restrict__ alpha, float* __restrict__ mbuf, int E) {
  int e = blockIdx.x * blockDim.x + threadIdx.x;
  if (e >= E) return;
  int s = src[e], d = dst[e];
  const v8h* qv = (const v8h*)(q + (size_t)d * HID_);
  const v8h* kv = (const v8h*)(kr + (size_t)s * HID_);
#pragma unroll
  for (int h = 0; h < H_; ++h) {
    float acc = 0.f;
#pragma unroll
    for (int i = 0; i < 4; ++i) {
      v8h a = qv[h * 4 + i], b = kv[h * 4 + i];
#pragma unroll
      for (int t = 0; t < 8; ++t) acc += (float)a[t] * (float)b[t];
    }
    float al = acc * prel[h] * kSCALE;
    alpha[(size_t)e * H_ + h] = al;
    atomicMaxF(&mbuf[(size_t)d * H_ + h], al);
  }
}

// pass B: alpha <- exp(alpha - m[dst]);  s = segsum(alpha)
__global__ void hgt_edge_expsum(const int* __restrict__ dst, float* __restrict__ alpha,
                                const float* __restrict__ mbuf, float* __restrict__ sbuf,
                                int E4) {
  int i = blockIdx.x * blockDim.x + threadIdx.x;
  if (i >= E4) return;
  int e = i >> 2, h = i & 3;
  int d = dst[e];
  float ex = __expf(alpha[i] - mbuf[(size_t)d * H_ + h]);
  alpha[i] = ex;
  atomicAdd(&sbuf[(size_t)d * H_ + h], ex);
}

// pass C: out[dst,h,:] += vr[src,h,:] * alpha/(s[dst,h]+eps)
__global__ void hgt_edge_scatter(const int* __restrict__ src, const int* __restrict__ dst,
                                 const float* __restrict__ alpha, const float* __restrict__ sbuf,
                                 const _Float16* __restrict__ vr, float* __restrict__ outp,
                                 int E4) {
  int i = blockIdx.x * blockDim.x + threadIdx.x;
  if (i >= E4) return;
  int e = i >> 2, h = i & 3;
  int s = src[e], d = dst[e];
  float w = alpha[i] / (sbuf[(size_t)d * H_ + h] + 1e-16f);
  const v8h* v = (const v8h*)(vr + (size_t)s * HID_ + h * 32);
  float* o = outp + (size_t)d * HID_ + h * 32;
#pragma unroll
  for (int c = 0; c < 4; ++c) {
    v8h x = v[c];
#pragma unroll
    for (int t = 0; t < 8; ++t) atomicAdd(&o[c * 8 + t], (float)x[t] * w);
  }
}

// ---------------------------------------------------------------- driver
extern "C" void kernel_launch(void* const* d_in, const int* in_sizes, int n_in,
                              void* d_out, int out_size, void* d_ws, size_t ws_size,
                              hipStream_t stream) {
  const float* x_paper      = (const float*)d_in[0];
  const float* x_author     = (const float*)d_in[1];
  const int*   wsrc         = (const int*)d_in[2];
  const int*   wdst         = (const int*)d_in[3];
  const int*   csrc         = (const int*)d_in[4];
  const int*   cdst         = (const int*)d_in[5];
  const float* lin_paper_w  = (const float*)d_in[6];
  const float* lin_paper_b  = (const float*)d_in[7];
  const float* lin_author_w = (const float*)d_in[8];
  const float* lin_author_b = (const float*)d_in[9];
  const float* lin_out_w    = (const float*)d_in[10];
  const float* lin_out_b    = (const float*)d_in[11];
  const float* k_w_paper    = (const float*)d_in[12];
  const float* k_b_paper    = (const float*)d_in[13];
  const float* q_w_paper    = (const float*)d_in[14];
  const float* q_b_paper    = (const float*)d_in[15];
  const float* v_w_paper    = (const float*)d_in[16];
  const float* v_b_paper    = (const float*)d_in[17];
  const float* a_w_paper    = (const float*)d_in[18];
  const float* a_b_paper    = (const float*)d_in[19];
  const float* skip_paper   = (const float*)d_in[20];
  const float* k_w_author   = (const float*)d_in[21];
  const float* k_b_author   = (const float*)d_in[22];
  const float* v_w_author   = (const float*)d_in[25];
  const float* v_b_author   = (const float*)d_in[26];
  const float* a_rel_writes = (const float*)d_in[30];
  const float* m_rel_writes = (const float*)d_in[31];
  const float* p_rel_writes = (const float*)d_in[32];
  const float* a_rel_cites  = (const float*)d_in[33];
  const float* m_rel_cites  = (const float*)d_in[34];
  const float* p_rel_cites  = (const float*)d_in[35];

  // ---- workspace carve-out (256B aligned chunks) ----
  char* wsp = (char*)d_ws;
  size_t off = 0;
  auto alloc = [&](size_t bytes) {
    void* p = (void*)(wsp + off);
    off += (bytes + 255) & ~(size_t)255;
    return p;
  };
  float*     hp     = (float*)alloc((size_t)NP_ * HID_ * 4);
  float*     ha     = (float*)alloc((size_t)NA_ * HID_ * 4);
  _Float16*  qp     = (_Float16*)alloc((size_t)NP_ * HID_ * 2);
  _Float16*  krw    = (_Float16*)alloc((size_t)NA_ * HID_ * 2);
  _Float16*  vrw    = (_Float16*)alloc((size_t)NA_ * HID_ * 2);
  _Float16*  krc    = (_Float16*)alloc((size_t)NP_ * HID_ * 2);
  _Float16*  vrc    = (_Float16*)alloc((size_t)NP_ * HID_ * 2);
  float*     outp   = (float*)alloc((size_t)NP_ * HID_ * 4);
  float*     hfin   = (float*)alloc((size_t)NP_ * HID_ * 4);
  float*     mbuf   = (float*)alloc((size_t)NP_ * H_ * 4);
  float*     sbuf   = (float*)alloc((size_t)NP_ * H_ * 4);
  float*     alphab = (float*)alloc((size_t)EW_ * H_ * 4);   // reused (EW_==EC_)
  _Float16*  bs_linp = (_Float16*)alloc(256 * 128 * 2);
  _Float16*  bs_lina = (_Float16*)alloc(128 * 128 * 2);
  _Float16*  bs_q    = (_Float16*)alloc(128 * 128 * 2);
  _Float16*  bs_aw   = (_Float16*)alloc(128 * 128 * 2);
  _Float16*  bs_out  = (_Float16*)alloc(128 * 16 * 2);
  _Float16*  bs_kw   = (_Float16*)alloc(128 * 128 * 2);  float* beff_kw = (float*)alloc(128 * 4);
  _Float16*  bs_vw   = (_Float16*)alloc(128 * 128 * 2);  float* beff_vw = (float*)alloc(128 * 4);
  _Float16*  bs_kc   = (_Float16*)alloc(128 * 128 * 2);  float* beff_kc = (float*)alloc(128 * 4);
  _Float16*  bs_vc   = (_Float16*)alloc(128 * 128 * 2);  float* beff_vc = (float*)alloc(128 * 4);

  // ---- 1) weight prep: swizzle to f16 WMMA layout; compose relation k/v ----
  hgt_prep_b<<<(256 * 128 + 255) / 256, 256, 0, stream>>>(lin_paper_w,  bs_linp, 256, 128);
  hgt_prep_b<<<(128 * 128 + 255) / 256, 256, 0, stream>>>(lin_author_w, bs_lina, 128, 128);
  hgt_prep_b<<<(128 * 128 + 255) / 256, 256, 0, stream>>>(q_w_paper,    bs_q,    128, 128);
  hgt_prep_b<<<(128 * 128 + 255) / 256, 256, 0, stream>>>(a_w_paper,    bs_aw,   128, 128);
  hgt_prep_b<<<(128 * 16  + 255) / 256, 256, 0, stream>>>(lin_out_w,    bs_out,  128, 16);
  int cgrid = (129 * 128 + 255) / 256;
  hgt_compose<<<cgrid, 256, 0, stream>>>(k_w_author, k_b_author, a_rel_writes, bs_kw, beff_kw);
  hgt_compose<<<cgrid, 256, 0, stream>>>(v_w_author, v_b_author, m_rel_writes, bs_vw, beff_vw);
  hgt_compose<<<cgrid, 256, 0, stream>>>(k_w_paper,  k_b_paper,  a_rel_cites,  bs_kc, beff_kc);
  hgt_compose<<<cgrid, 256, 0, stream>>>(v_w_paper,  v_b_paper,  m_rel_cites,  bs_vc, beff_vc);

  // ---- 2) dense projections via WMMA (LDS-free) ----
  dim3 gp((NP_ + 127) / 128), ga((NA_ + 127) / 128);
  hgt_gemm<0, 1, false, 8, float>    <<<gp, 256, 0, stream>>>(x_paper,  bs_linp, lin_paper_b,  nullptr, nullptr, hp,  NP_, 256);
  hgt_gemm<0, 1, false, 8, float>    <<<ga, 256, 0, stream>>>(x_author, bs_lina, lin_author_b, nullptr, nullptr, ha,  NA_, 128);
  hgt_gemm<0, 0, false, 8, _Float16> <<<gp, 256, 0, stream>>>(hp, bs_q,  q_b_paper, nullptr, nullptr, qp,  NP_, 128);
  hgt_gemm<0, 0, false, 8, _Float16> <<<ga, 256, 0, stream>>>(ha, bs_kw, beff_kw,   nullptr, nullptr, krw, NA_, 128);
  hgt_gemm<0, 0, false, 8, _Float16> <<<ga, 256, 0, stream>>>(ha, bs_vw, beff_vw,   nullptr, nullptr, vrw, NA_, 128);
  hgt_gemm<0, 0, false, 8, _Float16> <<<gp, 256, 0, stream>>>(hp, bs_kc, beff_kc,   nullptr, nullptr, krc, NP_, 128);
  hgt_gemm<0, 0, false, 8, _Float16> <<<gp, 256, 0, stream>>>(hp, bs_vc, beff_vc,   nullptr, nullptr, vrc, NP_, 128);

  // ---- 3) init accumulators ----
  int nOut = NP_ * HID_, nMH = NP_ * H_;
  hgt_fill<<<(nOut + 255) / 256, 256, 0, stream>>>(outp, 0.f, nOut);
  hgt_fill<<<(nMH + 255) / 256, 256, 0, stream>>>(mbuf, -3.0e38f, nMH);
  hgt_fill<<<(nMH + 255) / 256, 256, 0, stream>>>(sbuf, 0.f, nMH);

  // ---- 4) writes relation (author -> paper) ----
  int ge = (EW_ + 255) / 256, ge4 = (EW_ * 4 + 255) / 256;
  hgt_edge_alpha  <<<ge,  256, 0, stream>>>(qp, krw, wsrc, wdst, p_rel_writes, alphab, mbuf, EW_);
  hgt_edge_expsum <<<ge4, 256, 0, stream>>>(wdst, alphab, mbuf, sbuf, EW_ * 4);
  hgt_edge_scatter<<<ge4, 256, 0, stream>>>(wsrc, wdst, alphab, sbuf, vrw, outp, EW_ * 4);

  // ---- 5) cites relation (paper -> paper); reset per-relation softmax state ----
  hgt_fill<<<(nMH + 255) / 256, 256, 0, stream>>>(mbuf, -3.0e38f, nMH);
  hgt_fill<<<(nMH + 255) / 256, 256, 0, stream>>>(sbuf, 0.f, nMH);
  hgt_edge_alpha  <<<ge,  256, 0, stream>>>(qp, krc, csrc, cdst, p_rel_cites, alphab, mbuf, EC_);
  hgt_edge_expsum <<<ge4, 256, 0, stream>>>(cdst, alphab, mbuf, sbuf, EC_ * 4);
  hgt_edge_scatter<<<ge4, 256, 0, stream>>>(csrc, cdst, alphab, sbuf, vrc, outp, EC_ * 4);

  // ---- 6) output head: GELU -> a_w proj -> gated skip -> final linear ----
  hgt_gemm<1, 0, true, 8, float><<<gp, 256, 0, stream>>>(outp, bs_aw, a_b_paper, hp, skip_paper, hfin, NP_, 128);
  hgt_gemm<0, 0, false, 1, float><<<gp, 256, 0, stream>>>(hfin, bs_out, lin_out_b, nullptr, nullptr, (float*)d_out, NP_, 128);
}
